// Critic_Net_26345329394337
// MI455X (gfx1250) — matrix operand order repaired and checked by reading, not compiled, not run on hardware
//
#include <hip/hip_runtime.h>

#define Bsz 512
#define Tt 100
#define Ee 128
#define Hh 128
#define Dd 256
#define G4H 512
#define MIDn 100

typedef __bf16 bhalf;
typedef __attribute__((ext_vector_type(16))) __bf16 v16bf;
typedef __attribute__((ext_vector_type(8)))  float  v8f;
typedef __attribute__((ext_vector_type(4)))  float  vf4;

union V16U { v16bf v; vf4 f[2]; };

// ---- WMMA operand loaders (wave32, 16x16x32 bf16 layouts per CDNA5 ISA 7.12.2) ----

// A matrix tile (16 rows x 32 K), row-major source with leading dim `ld` (elements).
// lane<16: M=lane, K in [0,8)+[16,24); lane>=16: M=lane-16, K in [8,16)+[24,32)
__device__ __forceinline__ v16bf load_A(const bhalf* base, int ld, int lane) {
    const int m  = lane & 15;
    const int ks = (lane & 16) ? 8 : 0;
    const bhalf* p = base + (size_t)m * ld + ks;
    V16U u;
    u.f[0] = *reinterpret_cast<const vf4*>(p);
    u.f[1] = *reinterpret_cast<const vf4*>(p + 16);
    return u.v;
}

// B matrix tile (32 K x 16 cols) from pre-transposed weight WT[k][n], leading dim `ld`.
// lane l holds column N=l%16; K in [0,8)+[16,24) for lane<16, [8,16)+[24,32) for lane>=16.
__device__ __forceinline__ v16bf load_Bm(const bhalf* base, int ld, int lane) {
    const int n  = lane & 15;
    const int ks = (lane & 16) ? 8 : 0;
    v16bf v;
    #pragma unroll
    for (int s = 0; s < 8; ++s) {
        v[s]     = base[(size_t)(ks + s) * ld + n];
        v[s + 8] = base[(size_t)(16 + ks + s) * ld + n];
    }
    return v;
}

__device__ __forceinline__ v8f wmma_bf16(v16bf a, v16bf b, v8f c) {
    return __builtin_amdgcn_wmma_f32_16x16x32_bf16(false, a, false, b, (short)0, c,
                                                   false, false);
}

__device__ __forceinline__ float sigmoidf(float x) { return 1.0f / (1.0f + __expf(-x)); }

__device__ __forceinline__ float wave_sum(float v) {
    #pragma unroll
    for (int off = 16; off; off >>= 1) v += __shfl_xor(v, off, 32);
    return v;
}

// ---- software-pipelined 4-gate GEMM core for LSTM steps ----------------------
// gates[g] (+)= X[16 x KX] @ WihT[:, g*Hh+n0 : +16]  +  Hp[16 x KH] @ WhhT[...]
// One-chunk-ahead prefetch: next chunk's loads are in flight during current WMMAs.
template <int KX>
__device__ __forceinline__ void lstm_gates_wmma(
    const bhalf* __restrict__ x, int ldx,
    const bhalf* __restrict__ hp,
    const bhalf* __restrict__ WihT, const bhalf* __restrict__ WhhT,
    int n0, int lane,
    v8f& acc0, v8f& acc1, v8f& acc2, v8f& acc3) {
    constexpr int CX = KX / 32;
    constexpr int CH = Hh / 32;
    constexpr int NC = CX + CH;

    v16bf a_cur = load_A(x, ldx, lane);
    v16bf b0 = load_Bm(WihT + 0 * Hh + n0, G4H, lane);
    v16bf b1 = load_Bm(WihT + 1 * Hh + n0, G4H, lane);
    v16bf b2 = load_Bm(WihT + 2 * Hh + n0, G4H, lane);
    v16bf b3 = load_Bm(WihT + 3 * Hh + n0, G4H, lane);

    #pragma unroll
    for (int kk = 0; kk < NC; ++kk) {
        v16bf a_nxt = a_cur, n0b = b0, n1b = b1, n2b = b2, n3b = b3;
        if (kk + 1 < NC) {
            const int k1 = kk + 1;
            const bhalf* ap  = (k1 < CX) ? (x + k1 * 32) : (hp + (k1 - CX) * 32);
            const int    lda = (k1 < CX) ? ldx : Hh;
            const bhalf* W = (k1 < CX) ? (WihT + (size_t)(k1 * 32) * G4H)
                                       : (WhhT + (size_t)((k1 - CX) * 32) * G4H);
            a_nxt = load_A(ap, lda, lane);
            n0b = load_Bm(W + 0 * Hh + n0, G4H, lane);
            n1b = load_Bm(W + 1 * Hh + n0, G4H, lane);
            n2b = load_Bm(W + 2 * Hh + n0, G4H, lane);
            n3b = load_Bm(W + 3 * Hh + n0, G4H, lane);
        }
        acc0 = wmma_bf16(a_cur, b0, acc0);
        acc1 = wmma_bf16(a_cur, b1, acc1);
        acc2 = wmma_bf16(a_cur, b2, acc2);
        acc3 = wmma_bf16(a_cur, b3, acc3);
        a_cur = a_nxt; b0 = n0b; b1 = n1b; b2 = n2b; b3 = n3b;
    }
}

// ---------------- prep kernels ----------------

__global__ void embed_kernel(const float* __restrict__ inputs,
                             const float* __restrict__ W_embed,
                             bhalf* __restrict__ emb) {
    int id = blockIdx.x * blockDim.x + threadIdx.x;            // (t*Bsz + b)*Ee + e
    if (id >= Tt * Bsz * Ee) return;
    int e  = id % Ee;
    int tb = id / Ee;
    int b  = tb % Bsz;
    int t  = tb / Bsz;
    float x0 = inputs[((size_t)b * Tt + t) * 2 + 0];
    float x1 = inputs[((size_t)b * Tt + t) * 2 + 1];
    emb[id] = (bhalf)(x0 * W_embed[e * 2 + 0] + x1 * W_embed[e * 2 + 1]);
}

// dst[c*rows + r] = (bf16) src[r*cols + c]   (build WT[k][n] from W[n][k])
__global__ void transpose_bf16_kernel(const float* __restrict__ src,
                                      bhalf* __restrict__ dst, int rows, int cols) {
    int id = blockIdx.x * blockDim.x + threadIdx.x;
    if (id >= rows * cols) return;
    int c = id % cols, r = id / cols;
    dst[(size_t)c * rows + r] = (bhalf)src[id];
}

// ---------------- fused LSTM step kernels (WMMA + cell update) ----------------
// grid: (Bsz/16, 2 dirs), block: 256 (8 waves); wave w owns hidden cols [16w,16w+16)

__global__ void __launch_bounds__(256) lstm_enc_step(
    const bhalf* __restrict__ emb,
    const bhalf* __restrict__ WihT_f, const bhalf* __restrict__ WihT_b,
    const bhalf* __restrict__ WhhT_f, const bhalf* __restrict__ WhhT_b,
    const float* __restrict__ bias_f, const float* __restrict__ bias_b,
    const bhalf* __restrict__ hprev_f, const bhalf* __restrict__ hprev_b,
    bhalf* __restrict__ hnext_f, bhalf* __restrict__ hnext_b,
    float* __restrict__ cst_f, float* __restrict__ cst_b,
    bhalf* __restrict__ enc_out, int t) {
    const int lane = threadIdx.x & 31;
    const int wave = threadIdx.x >> 5;
    const int m0   = blockIdx.x * 16;
    const int dir  = blockIdx.y;
    const int n0   = wave * 16;

    const bhalf* WihT = dir ? WihT_b : WihT_f;
    const bhalf* WhhT = dir ? WhhT_b : WhhT_f;
    const float* bias = dir ? bias_b : bias_f;
    const bhalf* hp0  = dir ? hprev_b : hprev_f;
    bhalf*       hn   = dir ? hnext_b : hnext_f;
    float*       cst  = dir ? cst_b  : cst_f;
    const int xt = dir ? (Tt - 1 - t) : t;

    const bhalf* x  = emb + ((size_t)xt * Bsz + m0) * Ee;
    const bhalf* hp = hp0 + (size_t)m0 * Hh;

    v8f acc0 = {}, acc1 = {}, acc2 = {}, acc3 = {};
    lstm_gates_wmma<Ee>(x, Ee, hp, WihT, WhhT, n0, lane, acc0, acc1, acc2, acc3);

    const int n = n0 + (lane & 15);
    const float bi = bias[n], bff = bias[n + Hh], bg = bias[n + 2 * Hh], bo = bias[n + 3 * Hh];
    const int mbase = m0 + ((lane >> 4) << 3);
    #pragma unroll
    for (int r = 0; r < 8; ++r) {
        const int m = mbase + r;
        float gi = sigmoidf(acc0[r] + bi);
        float gf = sigmoidf(acc1[r] + bff);
        float gg = tanhf(acc2[r] + bg);
        float go = sigmoidf(acc3[r] + bo);
        float cp = cst[(size_t)m * Hh + n];
        float c2 = gf * cp + gi * gg;
        float h2 = go * tanhf(c2);
        cst[(size_t)m * Hh + n] = c2;
        hn[(size_t)m * Hh + n]  = (bhalf)h2;
        enc_out[((size_t)m * Tt + xt) * Dd + dir * Hh + n] = (bhalf)h2;
    }
}

__global__ void __launch_bounds__(256) lstm_proc_step(
    const bhalf* __restrict__ pin,
    const bhalf* __restrict__ WihT_f, const bhalf* __restrict__ WihT_b,
    const bhalf* __restrict__ WhhT_f, const bhalf* __restrict__ WhhT_b,
    const float* __restrict__ bias_f, const float* __restrict__ bias_b,
    const bhalf* __restrict__ hprev_f, const bhalf* __restrict__ hprev_b,
    bhalf* __restrict__ hnext_f, bhalf* __restrict__ hnext_b,
    float* __restrict__ cst_f, float* __restrict__ cst_b,
    float* __restrict__ pout, bhalf* __restrict__ pout_bf) {
    const int lane = threadIdx.x & 31;
    const int wave = threadIdx.x >> 5;
    const int m0   = blockIdx.x * 16;
    const int dir  = blockIdx.y;
    const int n0   = wave * 16;

    const bhalf* WihT = dir ? WihT_b : WihT_f;
    const bhalf* WhhT = dir ? WhhT_b : WhhT_f;
    const float* bias = dir ? bias_b : bias_f;
    const bhalf* hp0  = dir ? hprev_b : hprev_f;
    bhalf*       hn   = dir ? hnext_b : hnext_f;
    float*       cst  = dir ? cst_b  : cst_f;

    const bhalf* x  = pin + (size_t)m0 * Dd;
    const bhalf* hp = hp0 + (size_t)m0 * Hh;

    v8f acc0 = {}, acc1 = {}, acc2 = {}, acc3 = {};
    lstm_gates_wmma<Dd>(x, Dd, hp, WihT, WhhT, n0, lane, acc0, acc1, acc2, acc3);

    const int n = n0 + (lane & 15);
    const float bi = bias[n], bff = bias[n + Hh], bg = bias[n + 2 * Hh], bo = bias[n + 3 * Hh];
    const int mbase = m0 + ((lane >> 4) << 3);
    #pragma unroll
    for (int r = 0; r < 8; ++r) {
        const int m = mbase + r;
        float gi = sigmoidf(acc0[r] + bi);
        float gf = sigmoidf(acc1[r] + bff);
        float gg = tanhf(acc2[r] + bg);
        float go = sigmoidf(acc3[r] + bo);
        float cp = cst[(size_t)m * Hh + n];
        float c2 = gf * cp + gi * gg;
        float h2 = go * tanhf(c2);
        cst[(size_t)m * Hh + n] = c2;
        hn[(size_t)m * Hh + n]  = (bhalf)h2;
        pout[(size_t)m * Dd + dir * Hh + n]    = h2;       // concat(shf, shb)
        pout_bf[(size_t)m * Dd + dir * Hh + n] = (bhalf)h2;
    }
}

// ---------------- generic WMMA GEMM: C[f32] = A[bf16] @ Bt[bf16] ----------------
// grid: (M/16, N/128), block 256; wave w -> cols [(by*8+w)*16, +16)
// One-chunk-ahead prefetch in K.
__global__ void __launch_bounds__(256) gemm_bf16f32(
    const bhalf* __restrict__ A, int lda,
    const bhalf* __restrict__ Bt, int ldb,
    float* __restrict__ C, int ldc, int K) {
    const int lane = threadIdx.x & 31;
    const int wave = threadIdx.x >> 5;
    const int m0 = blockIdx.x * 16;
    const int n0 = (blockIdx.y * 8 + wave) * 16;
    const bhalf* a0 = A + (size_t)m0 * lda;
    v8f acc = {};
    v16bf a_cur = load_A(a0, lda, lane);
    v16bf b_cur = load_Bm(Bt + n0, ldb, lane);
    for (int kc = 0; kc < K; kc += 32) {
        v16bf a_nxt = a_cur, b_nxt = b_cur;
        if (kc + 32 < K) {
            a_nxt = load_A(a0 + kc + 32, lda, lane);
            b_nxt = load_Bm(Bt + (size_t)(kc + 32) * ldb + n0, ldb, lane);
        }
        acc = wmma_bf16(a_cur, b_cur, acc);
        a_cur = a_nxt; b_cur = b_nxt;
    }
    const int n = n0 + (lane & 15);
    const int mbase = m0 + ((lane >> 4) << 3);
    #pragma unroll
    for (int r = 0; r < 8; ++r)
        C[(size_t)(mbase + r) * ldc + n] = acc[r];
}

// ---------------- attention kernels ----------------

// one wave per (b,t): s = v . tanh(ref[b,t,:] + q[b,:]) + v_b
__global__ void __launch_bounds__(256) scores_kernel(
    const float* __restrict__ ref, const float* __restrict__ q,
    const float* __restrict__ v_w, const float* __restrict__ v_b,
    float* __restrict__ scores) {
    const int lane = threadIdx.x & 31;
    const int gw = blockIdx.x * 8 + (threadIdx.x >> 5);
    const int b = gw / Tt, t = gw % Tt;
    const vf4* rp = reinterpret_cast<const vf4*>(ref + ((size_t)b * Tt + t) * Dd + lane * 8);
    const vf4* qp = reinterpret_cast<const vf4*>(q + (size_t)b * Dd + lane * 8);
    const vf4* vp = reinterpret_cast<const vf4*>(v_w + lane * 8);
    float s = 0.f;
    #pragma unroll
    for (int i = 0; i < 2; ++i) {
        vf4 r4 = rp[i], q4 = qp[i], v4 = vp[i];
        s += v4.x * tanhf(r4.x + q4.x);
        s += v4.y * tanhf(r4.y + q4.y);
        s += v4.z * tanhf(r4.z + q4.z);
        s += v4.w * tanhf(r4.w + q4.w);
    }
    s = wave_sum(s);
    if (lane == 0) scores[(size_t)b * Tt + t] = s + v_b[0];
}

// one wave per batch row: softmax over T=100
__global__ void __launch_bounds__(256) softmax_kernel(
    const float* __restrict__ scores, float* __restrict__ attn) {
    const int lane = threadIdx.x & 31;
    const int b = blockIdx.x * 8 + (threadIdx.x >> 5);
    float v[4];
    float mx = -1e30f;
    #pragma unroll
    for (int i = 0; i < 4; ++i) {
        int t = lane + i * 32;
        v[i] = (t < Tt) ? scores[(size_t)b * Tt + t] : -1e30f;
        mx = fmaxf(mx, v[i]);
    }
    #pragma unroll
    for (int off = 16; off; off >>= 1) mx = fmaxf(mx, __shfl_xor(mx, off, 32));
    float sum = 0.f;
    #pragma unroll
    for (int i = 0; i < 4; ++i) {
        int t = lane + i * 32;
        v[i] = (t < Tt) ? __expf(v[i] - mx) : 0.f;
        sum += v[i];
    }
    sum = wave_sum(sum);
    const float inv = 1.0f / sum;
    #pragma unroll
    for (int i = 0; i < 4; ++i) {
        int t = lane + i * 32;
        if (t < Tt) attn[(size_t)b * Tt + t] = v[i] * inv;
    }
}

// pin[b,d] = sum_t attn[b,t] * enc[b,t,d]   (enc is L2-resident bf16)
__global__ void __launch_bounds__(512) glimpse_kernel(
    const float* __restrict__ attn, const bhalf* __restrict__ enc,
    bhalf* __restrict__ pin) {
    const int id = blockIdx.x * blockDim.x + threadIdx.x;   // b*Dd + d
    const int b = id >> 8;
    const int d = id & (Dd - 1);
    const bhalf* e = enc + (size_t)b * Tt * Dd + d;
    const float* a = attn + (size_t)b * Tt;
    float s = 0.f;
    #pragma unroll 4
    for (int t = 0; t < Tt; ++t) s += a[t] * (float)e[(size_t)t * Dd];
    pin[id] = (bhalf)s;
}

// out[b] = W2 . relu(W1 @ pout[b] + b1) + b2
__global__ void __launch_bounds__(128) mlp_kernel(
    const float* __restrict__ pout, const float* __restrict__ W1,
    const float* __restrict__ b1, const float* __restrict__ W2,
    const float* __restrict__ b2, float* __restrict__ out) {
    __shared__ float sh[128];
    const int b = blockIdx.x, j = threadIdx.x;
    float val = 0.f;
    if (j < MIDn) {
        const float* p = pout + (size_t)b * Dd;
        const float* w = W1 + (size_t)j * Dd;
        float s = b1[j];
        #pragma unroll 4
        for (int d = 0; d < Dd; ++d) s += p[d] * w[d];
        val = fmaxf(s, 0.f) * W2[j];
    }
    sh[j] = val;
    __syncthreads();
    if (j == 0) {
        float s = b2[0];
        for (int k = 0; k < 128; ++k) s += sh[k];
        out[b] = s;
    }
}

// ---------------- host launcher ----------------

extern "C" void kernel_launch(void* const* d_in, const int* in_sizes, int n_in,
                              void* d_out, int out_size, void* d_ws, size_t ws_size,
                              hipStream_t stream) {
    const float* inputs     = (const float*)d_in[0];
    const float* W_embed    = (const float*)d_in[1];
    const float* enc_Wih_f  = (const float*)d_in[2];
    const float* enc_Whh_f  = (const float*)d_in[3];
    const float* enc_b_f    = (const float*)d_in[4];
    const float* enc_Wih_b  = (const float*)d_in[5];
    const float* enc_Whh_b  = (const float*)d_in[6];
    const float* enc_b_b    = (const float*)d_in[7];
    const float* W_ref      = (const float*)d_in[8];
    const float* W_q        = (const float*)d_in[9];
    const float* v_w        = (const float*)d_in[10];
    const float* v_b        = (const float*)d_in[11];
    const float* proc_Wih_f = (const float*)d_in[12];
    const float* proc_Whh_f = (const float*)d_in[13];
    const float* proc_b_f   = (const float*)d_in[14];
    const float* proc_Wih_b = (const float*)d_in[15];
    const float* proc_Whh_b = (const float*)d_in[16];
    const float* proc_b_b   = (const float*)d_in[17];
    const float* W1         = (const float*)d_in[18];
    const float* b1         = (const float*)d_in[19];
    const float* W2         = (const float*)d_in[20];
    const float* b2         = (const float*)d_in[21];

    char* ws = (char*)d_ws;
    size_t off = 0;
    auto alloc = [&](size_t bytes) -> char* {
        char* p = ws + off;
        off += (bytes + 255) & ~(size_t)255;
        return p;
    };

    bhalf* emb      = (bhalf*)alloc((size_t)Tt * Bsz * Ee * 2);
    bhalf* eWihT_f  = (bhalf*)alloc((size_t)Ee * G4H * 2);
    bhalf* eWihT_b  = (bhalf*)alloc((size_t)Ee * G4H * 2);
    bhalf* eWhhT_f  = (bhalf*)alloc((size_t)Hh * G4H * 2);
    bhalf* eWhhT_b  = (bhalf*)alloc((size_t)Hh * G4H * 2);
    bhalf* pWihT_f  = (bhalf*)alloc((size_t)Dd * G4H * 2);
    bhalf* pWihT_b  = (bhalf*)alloc((size_t)Dd * G4H * 2);
    bhalf* pWhhT_f  = (bhalf*)alloc((size_t)Hh * G4H * 2);
    bhalf* pWhhT_b  = (bhalf*)alloc((size_t)Hh * G4H * 2);
    bhalf* WrefT    = (bhalf*)alloc((size_t)Dd * Dd * 2);
    bhalf* WqT      = (bhalf*)alloc((size_t)Dd * Dd * 2);
    bhalf* hbuf_f[2] = {(bhalf*)alloc((size_t)Bsz * Hh * 2), (bhalf*)alloc((size_t)Bsz * Hh * 2)};
    bhalf* hbuf_b[2] = {(bhalf*)alloc((size_t)Bsz * Hh * 2), (bhalf*)alloc((size_t)Bsz * Hh * 2)};
    float* c_f      = (float*)alloc((size_t)Bsz * Hh * 4);
    float* c_b      = (float*)alloc((size_t)Bsz * Hh * 4);
    bhalf* enc      = (bhalf*)alloc((size_t)Bsz * Tt * Dd * 2);
    float* ref      = (float*)alloc((size_t)Bsz * Tt * Dd * 4);
    bhalf* pin      = (bhalf*)alloc((size_t)Bsz * Dd * 2);
    float* pout     = (float*)alloc((size_t)Bsz * Dd * 4);
    bhalf* pout_bf  = (bhalf*)alloc((size_t)Bsz * Dd * 2);
    float* qbuf     = (float*)alloc((size_t)Bsz * Dd * 4);
    float* scores   = (float*)alloc((size_t)Bsz * Tt * 4);
    float* attn     = (float*)alloc((size_t)Bsz * Tt * 4);

    // --- prep: embedding + bf16 transposed weights + zero-init states ---
    {
        int n = Tt * Bsz * Ee;
        embed_kernel<<<(n + 255) / 256, 256, 0, stream>>>(inputs, W_embed, emb);
    }
    auto tr = [&](const float* s, bhalf* d, int rows, int cols) {
        int n = rows * cols;
        transpose_bf16_kernel<<<(n + 255) / 256, 256, 0, stream>>>(s, d, rows, cols);
    };
    tr(enc_Wih_f,  eWihT_f, G4H, Ee);
    tr(enc_Whh_f,  eWhhT_f, G4H, Hh);
    tr(enc_Wih_b,  eWihT_b, G4H, Ee);
    tr(enc_Whh_b,  eWhhT_b, G4H, Hh);
    tr(proc_Wih_f, pWihT_f, G4H, Dd);
    tr(proc_Whh_f, pWhhT_f, G4H, Hh);
    tr(proc_Wih_b, pWihT_b, G4H, Dd);
    tr(proc_Whh_b, pWhhT_b, G4H, Hh);
    tr(W_ref, WrefT, Dd, Dd);
    tr(W_q,   WqT,   Dd, Dd);

    hipMemsetAsync(hbuf_f[0], 0, (size_t)Bsz * Hh * 2, stream);
    hipMemsetAsync(hbuf_b[0], 0, (size_t)Bsz * Hh * 2, stream);
    hipMemsetAsync(c_f, 0, (size_t)Bsz * Hh * 4, stream);
    hipMemsetAsync(c_b, 0, (size_t)Bsz * Hh * 4, stream);
    hipMemsetAsync(pin, 0, (size_t)Bsz * Dd * 2, stream);

    // --- bidirectional encoder LSTM: 100 sequential fused WMMA steps ---
    for (int t = 0; t < Tt; ++t) {
        lstm_enc_step<<<dim3(Bsz / 16, 2), 256, 0, stream>>>(
            emb, eWihT_f, eWihT_b, eWhhT_f, eWhhT_b, enc_b_f, enc_b_b,
            hbuf_f[t & 1], hbuf_b[t & 1], hbuf_f[(t + 1) & 1], hbuf_b[(t + 1) & 1],
            c_f, c_b, enc, t);
    }

    // --- ref = enc @ W_ref^T  ([51200,256]x[256,256]) ---
    gemm_bf16f32<<<dim3((Bsz * Tt) / 16, Dd / 128), 256, 0, stream>>>(
        enc, Dd, WrefT, Dd, ref, Dd, Dd);

    // --- process loop: cell -> q -> scores -> softmax -> glimpse (x100) ---
    // encoder final states (hbuf parity 0, c in place) seed the process cells.
    for (int s = 0; s < Tt; ++s) {
        lstm_proc_step<<<dim3(Bsz / 16, 2), 256, 0, stream>>>(
            pin, pWihT_f, pWihT_b, pWhhT_f, pWhhT_b, proc_b_f, proc_b_b,
            hbuf_f[s & 1], hbuf_b[s & 1], hbuf_f[(s + 1) & 1], hbuf_b[(s + 1) & 1],
            c_f, c_b, pout, pout_bf);
        gemm_bf16f32<<<dim3(Bsz / 16, Dd / 128), 256, 0, stream>>>(
            pout_bf, Dd, WqT, Dd, qbuf, Dd, Dd);
        scores_kernel<<<(Bsz * Tt) / 8, 256, 0, stream>>>(ref, qbuf, v_w, v_b, scores);
        softmax_kernel<<<Bsz / 8, 256, 0, stream>>>(scores, attn);
        glimpse_kernel<<<(Bsz * Dd) / 512, 512, 0, stream>>>(attn, enc, pin);
    }

    // --- final MLP on last pout ---
    mlp_kernel<<<Bsz, 128, 0, stream>>>(pout, W1, b1, W2, b2, (float*)d_out);
}
